// DARTSModel_57887569215900
// MI455X (gfx1250) — compile-verified, matrix operand order checked by reading, not run on hardware
//
#include <hip/hip_runtime.h>

typedef __bf16 bf16;
typedef __bf16 v8bf  __attribute__((ext_vector_type(8)));
typedef __bf16 v16bf __attribute__((ext_vector_type(16)));
typedef float  v8f   __attribute__((ext_vector_type(8)));

#define ACT_TANH 0
#define ACT_RELU 1
#define ACT_ID   2
#define ACT_SIG  3

// ---------------------------------------------------------------------------
// Global-memory fragment loader for 16-bit 16x32 A/B tiles (CDNA5 WMMA VGPR
// layout): lanes 0-15 hold row lane, K chunks [0..7]/[16..23]; lanes 16-31
// hold row lane-16, K chunks [8..15]/[24..31]. K must be contiguous.
// ---------------------------------------------------------------------------
__device__ __forceinline__ v16bf load_frag_g(const bf16* __restrict__ p0, int ld) {
  const int lane = threadIdx.x & 31;
  const bf16* p = p0 + (lane & 15) * ld + ((lane >> 4) << 3);
  v8bf lo = *reinterpret_cast<const v8bf*>(p);
  v8bf hi = *reinterpret_cast<const v8bf*>(p + 16);
  v16bf f;
#pragma unroll
  for (int i = 0; i < 8; ++i) { f[i] = lo[i]; f[i + 8] = hi[i]; }
  return f;
}

// ---------------------------------------------------------------------------
// LDS B-tile fragment loader. LDS holds frag-ordered data: 16-byte block for
// (k-chunk c, half h, lane l) lives at bf16 offset ((c*2+h)*32+l)*8. Each
// lane reads 2x16B; bank index = (lane*4)%64 -> 2-phase, conflict-minimal.
// ---------------------------------------------------------------------------
__device__ __forceinline__ v16bf lds_frag(const bf16* base, int c) {
  const int lane = threadIdx.x & 31;
  v8bf lo = *reinterpret_cast<const v8bf*>(base + ((c * 2 + 0) * 32 + lane) * 8);
  v8bf hi = *reinterpret_cast<const v8bf*>(base + ((c * 2 + 1) * 32 + lane) * 8);
  v16bf f;
#pragma unroll
  for (int i = 0; i < 8; ++i) { f[i] = lo[i]; f[i + 8] = hi[i]; }
  return f;
}

__device__ __forceinline__ v8f wmma_bf16(v16bf a, v16bf b, v8f c) {
  return __builtin_amdgcn_wmma_f32_16x16x32_bf16(false, a, false, b,
                                                 (short)0, c, false, false);
}

__device__ __forceinline__ float act_apply(int act, float x) {
  switch (act) {
    case ACT_TANH: return tanhf(x);
    case ACT_RELU: return fmaxf(x, 0.0f);
    case ACT_SIG:  return 1.0f / (1.0f + __expf(-x));
    default:       return x;
  }
}

// Stage both c-part (cols n0..n0+15) and g-part (cols n0+512..) of the
// transposed weights (1024x512, N-major) into LDS in frag order. 32KB total.
__device__ __forceinline__ void stage_B(const bf16* __restrict__ Wt, int n0,
                                        bf16* ldsB) {
#pragma unroll
  for (int j = 0; j < 8; ++j) {
    const int i = threadIdx.x + j * 256;       // 16B block index, 0..2047
    const int part = i >> 10;                  // 0 = c tile, 1 = g tile
    const int idx  = i & 1023;
    const int l  = idx & 31;
    const int ch = idx >> 5;                   // c*2 + h
    const int row = n0 + part * 512 + (l & 15);
    const int k   = (ch >> 1) * 32 + ((l >> 4) << 3) + (ch & 1) * 16;
    *reinterpret_cast<v8bf*>(ldsB + i * 8) =
        *reinterpret_cast<const v8bf*>(Wt + (size_t)row * 512 + k);
  }
  __syncthreads();
}

// Software grid barrier (counter + generation), agent scope so state writes
// are published to all blocks via the release->acquire chain.
__device__ __forceinline__ void grid_sync(unsigned* cnt, unsigned* gen,
                                          unsigned nblk) {
  __syncthreads();
  if (threadIdx.x == 0) {
    const unsigned g = __hip_atomic_load(gen, __ATOMIC_RELAXED,
                                         __HIP_MEMORY_SCOPE_AGENT);
    const unsigned old = __hip_atomic_fetch_add(cnt, 1u, __ATOMIC_ACQ_REL,
                                                __HIP_MEMORY_SCOPE_AGENT);
    if (old == nblk - 1u) {
      __hip_atomic_store(cnt, 0u, __ATOMIC_RELAXED, __HIP_MEMORY_SCOPE_AGENT);
      __hip_atomic_store(gen, g + 1u, __ATOMIC_RELEASE, __HIP_MEMORY_SCOPE_AGENT);
    } else {
      while (__hip_atomic_load(gen, __ATOMIC_ACQUIRE,
                               __HIP_MEMORY_SCOPE_AGENT) == g) {
        __builtin_amdgcn_s_sleep(2);
      }
    }
  }
  __syncthreads();
}

// GEMM + gate for one 16x16 tile: out = inp + sig(c) * (act(g) - inp)
__device__ __forceinline__ void gate_tile(const bf16* __restrict__ inp,
                                          const bf16* ldsB,
                                          bf16* __restrict__ outp, int act,
                                          int m0, int n0) {
  const int lane = threadIdx.x & 31;
  v8f accC = {}, accG = {};
  const bf16* A = inp + m0 * 512;
#pragma unroll 4
  for (int c = 0; c < 16; ++c) {
    v16bf a = load_frag_g(A + c * 32, 512);
    accC = wmma_bf16(a, lds_frag(ldsB, c), accC);
    accG = wmma_bf16(a, lds_frag(ldsB + 8192, c), accG);
  }
  const int n  = n0 + (lane & 15);
  const int mb = m0 + ((lane >> 4) << 3);
#pragma unroll
  for (int v = 0; v < 8; ++v) {
    const int idx = (mb + v) * 512 + n;
    float iv = (float)inp[idx];
    float sg = 1.0f / (1.0f + __expf(-accC[v]));
    outp[idx] = (bf16)(iv + sg * (act_apply(act, accG[v]) - iv));
  }
}

// ---------------------------------------------------------------------------
// Persistent recurrence kernel: 96 blocks x 256 threads walk all 256
// timesteps; 5 dependency levels per step separated by grid_sync. Per block,
// the 8 waves share one n-tile (n0) so B weights are staged once into LDS.
//   wave -> m-tile, blockIdx&31 -> n-tile, blockIdx>>5 -> gemm within level.
// ---------------------------------------------------------------------------
__global__ void __launch_bounds__(256)
recurrence_kernel(const bf16* __restrict__ Xc, const bf16* __restrict__ W0hT,
                  const bf16* __restrict__ WsT, const float* __restrict__ masks,
                  bf16* __restrict__ h, bf16* __restrict__ Sb,
                  float* __restrict__ out, unsigned* __restrict__ sync) {
  __shared__ bf16 ldsB[16384];                 // 32KB: c tile, then g tile
  const int lane = threadIdx.x & 31;
  const int m0 = (threadIdx.x >> 5) << 4;      // wave -> m-tile
  const int n0 = (blockIdx.x & 31) << 4;       // block -> n-tile
  const int g  = blockIdx.x >> 5;              // gemm index within level
  const int n  = n0 + (lane & 15);
  const int mb = m0 + ((lane >> 4) << 3);
  bf16* S[8];
#pragma unroll
  for (int i = 0; i < 8; ++i) S[i] = Sb + (size_t)i * 65536;
  const bf16* W[8];
#pragma unroll
  for (int i = 0; i < 8; ++i) W[i] = WsT + (size_t)i * 524288;
  unsigned* cnt = sync;
  unsigned* gen = sync + 1;

  for (int t = 0; t < 256; ++t) {
    // ---- L0: s0 = h + sig(c)*(tanh(g)-h), js = Xc[:,t,:] + h@W0h ----
    if (blockIdx.x < 32) {
      stage_B(W0hT, n0, ldsB);
      v8f accC, accG;
#pragma unroll
      for (int v = 0; v < 8; ++v) {
        const int row = ((mb + v) * 256 + t) * 1024;
        accC[v] = (float)Xc[row + n];
        accG[v] = (float)Xc[row + n + 512];
      }
      const bf16* A = h + m0 * 512;
#pragma unroll 4
      for (int c = 0; c < 16; ++c) {
        v16bf a = load_frag_g(A + c * 32, 512);
        accC = wmma_bf16(a, lds_frag(ldsB, c), accC);
        accG = wmma_bf16(a, lds_frag(ldsB + 8192, c), accG);
      }
#pragma unroll
      for (int v = 0; v < 8; ++v) {
        const int idx = (mb + v) * 512 + n;
        float hv = (float)h[idx];
        float sg = 1.0f / (1.0f + __expf(-accC[v]));
        S[0][idx] = (bf16)(hv + sg * (tanhf(accG[v]) - hv));
      }
    }
    grid_sync(cnt, gen, 96);

    // ---- L1: s1 = tanh-gate(s0@Ws0), s4 = relu-gate(s0@Ws3) ----
    if (blockIdx.x < 64) {
      const bf16* Wt = g ? W[3] : W[0];
      stage_B(Wt, n0, ldsB);
      gate_tile(S[0], ldsB, g ? S[4] : S[1], g ? ACT_RELU : ACT_TANH, m0, n0);
    }
    grid_sync(cnt, gen, 96);

    // ---- L2: s2 = relu(s1@Ws1), s3 = tanh(s1@Ws2), s7 = tanh(s4@Ws6) ----
    {
      const bf16* Wt  = (g == 0) ? W[1] : ((g == 1) ? W[2] : W[6]);
      const bf16* inp = (g == 2) ? S[4] : S[1];
      bf16* outp      = (g == 0) ? S[2] : ((g == 1) ? S[3] : S[7]);
      const int act   = (g == 0) ? ACT_RELU : ACT_TANH;
      stage_B(Wt, n0, ldsB);
      gate_tile(inp, ldsB, outp, act, m0, n0);
    }
    grid_sync(cnt, gen, 96);

    // ---- L3: s5 = id(s2@Ws4), s6 = sig(s3@Ws5) ----
    if (blockIdx.x < 64) {
      const bf16* Wt = g ? W[5] : W[4];
      stage_B(Wt, n0, ldsB);
      gate_tile(g ? S[3] : S[2], ldsB, g ? S[6] : S[5], g ? ACT_SIG : ACT_ID,
                m0, n0);
    }
    grid_sync(cnt, gen, 96);

    // ---- L4: s8 = relu-gate(s5@Ws7) fused with h = mean(s1..s8) ----
    if (blockIdx.x < 32) {
      stage_B(W[7], n0, ldsB);
      v8f accC = {}, accG = {};
      const bf16* A = S[5] + m0 * 512;
#pragma unroll 4
      for (int c = 0; c < 16; ++c) {
        v16bf a = load_frag_g(A + c * 32, 512);
        accC = wmma_bf16(a, lds_frag(ldsB, c), accC);
        accG = wmma_bf16(a, lds_frag(ldsB + 8192, c), accG);
      }
#pragma unroll
      for (int v = 0; v < 8; ++v) {
        const int m = mb + v;
        const int idx = m * 512 + n;
        float i5 = (float)S[5][idx];
        float sg = 1.0f / (1.0f + __expf(-accC[v]));
        float s8 = i5 + sg * (fmaxf(accG[v], 0.0f) - i5);
        float sum = (float)S[1][idx] + (float)S[2][idx] + (float)S[3][idx] +
                    (float)S[4][idx] + i5 + (float)S[6][idx] +
                    (float)S[7][idx] + s8;
        float hn = sum * 0.125f;
        h[idx] = (bf16)hn;
        out[((size_t)m * 256 + t) * 512 + n] = hn * masks[m * 256 + t];
      }
    }
    grid_sync(cnt, gen, 96);   // publish h before next step's L0
  }
}

// ---------------------------------------------------------------------------
// Prep kernels (one-time, off critical path)
// ---------------------------------------------------------------------------
__global__ void prep_transpose_kernel(const float* __restrict__ src, int src_ld,
                                      int row_off, int Kvalid, int Nvalid,
                                      bf16* __restrict__ dst, int Kp, int Np) {
  const int total = Np * Kp;
  for (int i = blockIdx.x * blockDim.x + threadIdx.x; i < total;
       i += gridDim.x * blockDim.x) {
    const int nn = i / Kp;
    const int k  = i - nn * Kp;
    float v = 0.0f;
    if (nn < Nvalid && k < Kvalid)
      v = src[(size_t)(row_off + k) * src_ld + nn];
    dst[i] = (bf16)v;
  }
}

__global__ void convert_pad_kernel(const float* __restrict__ src,
                                   bf16* __restrict__ dst,
                                   int rows, int cols, int colsp) {
  const int total = rows * colsp;
  for (int i = blockIdx.x * blockDim.x + threadIdx.x; i < total;
       i += gridDim.x * blockDim.x) {
    const int r = i / colsp;
    const int c = i - r * colsp;
    dst[i] = (bf16)((c < cols) ? src[(size_t)r * cols + c] : 0.0f);
  }
}

__global__ void zero_bf16_kernel(bf16* __restrict__ p, int n) {
  for (int i = blockIdx.x * blockDim.x + threadIdx.x; i < n;
       i += gridDim.x * blockDim.x)
    p[i] = (bf16)0.0f;
}

__global__ void init_sync_kernel(unsigned* __restrict__ p) {
  p[0] = 0u;
  p[1] = 0u;
}

// ---------------------------------------------------------------------------
// Big one-shot bf16 GEMM (encoder + Xc precompute): C = A * Wt^T (+bias)
// ---------------------------------------------------------------------------
__global__ void __launch_bounds__(256)
gemm_bf16_kernel(const bf16* __restrict__ A, const bf16* __restrict__ Wt,
                 const float* __restrict__ bias, bf16* __restrict__ C,
                 int M, int N, int K, int nbias) {
  const int lane   = threadIdx.x & 31;
  const int tile   = blockIdx.x * 8 + (threadIdx.x >> 5);
  const int ntiles = N >> 4;
  const int mt = tile / ntiles;
  const int nt = tile - mt * ntiles;
  if (mt * 16 >= M) return;                    // wave-uniform exit
  v8f acc = {};
  const bf16* Ap = A + (size_t)mt * 16 * K;
  const bf16* Bp = Wt + (size_t)nt * 16 * K;
#pragma unroll 2
  for (int k = 0; k < K; k += 32) {
    v16bf a = load_frag_g(Ap + k, K);
    v16bf b = load_frag_g(Bp + k, K);
    acc = wmma_bf16(a, b, acc);
  }
  const int n  = nt * 16 + (lane & 15);
  const int mb = mt * 16 + ((lane >> 4) << 3);
  const float bv = (bias != nullptr && n < nbias) ? bias[n] : 0.0f;
#pragma unroll
  for (int v = 0; v < 8; ++v)
    C[(size_t)(mb + v) * N + n] = (bf16)(acc[v] + bv);
}

// ---------------------------------------------------------------------------
// Host launcher
// ---------------------------------------------------------------------------
extern "C" void kernel_launch(void* const* d_in, const int* in_sizes, int n_in,
                              void* d_out, int out_size, void* d_ws, size_t ws_size,
                              hipStream_t stream) {
  (void)in_sizes; (void)n_in; (void)out_size; (void)ws_size;
  const float* inputs = (const float*)d_in[0];   // (128,256,360)
  // d_in[1] = hidden (zeros; reference ignores it too)
  const float* masks  = (const float*)d_in[2];   // (128,256)
  const float* W_enc  = (const float*)d_in[3];   // (360,300)
  const float* b_enc  = (const float*)d_in[4];   // (300,)
  const float* W0     = (const float*)d_in[5];   // (812,1024)
  const float* Ws     = (const float*)d_in[6];   // (8,512,1024)
  float* out = (float*)d_out;                    // (128,256,512) f32
  char* ws = (char*)d_ws;

  // Workspace layout. Abf (padded bf16 inputs) is dead before Xc is written,
  // so it overlaps the Xc region.
  bf16* Xc  = (bf16*)ws;                               // 32768x1024 bf16 (67MB)
  bf16* Abf = (bf16*)ws;                               // 32768x384  bf16 (25MB)
  size_t off = (size_t)32768 * 1024 * 2;
  bf16* X      = (bf16*)(ws + off); off += (size_t)32768 * 320 * 2;
  bf16* WencT  = (bf16*)(ws + off); off += (size_t)320 * 384 * 2;
  bf16* W0topT = (bf16*)(ws + off); off += (size_t)1024 * 320 * 2;
  bf16* W0hT   = (bf16*)(ws + off); off += (size_t)1024 * 512 * 2;
  bf16* WsT    = (bf16*)(ws + off); off += (size_t)8 * 1024 * 512 * 2;
  bf16* h      = (bf16*)(ws + off); off += (size_t)128 * 512 * 2;
  bf16* Sb     = (bf16*)(ws + off); off += (size_t)8 * 128 * 512 * 2;
  unsigned* syncp = (unsigned*)(ws + ((off + 255) & ~(size_t)255));

  // --- one-time prep: bf16 transposed weights, padded bf16 inputs, h0=0 ---
  prep_transpose_kernel<<<480, 256, 0, stream>>>(W_enc, 300, 0, 360, 300, WencT, 384, 320);
  prep_transpose_kernel<<<1280, 256, 0, stream>>>(W0, 1024, 0, 300, 1024, W0topT, 320, 1024);
  prep_transpose_kernel<<<2048, 256, 0, stream>>>(W0, 1024, 300, 512, 1024, W0hT, 512, 1024);
  for (int i = 0; i < 8; ++i)
    prep_transpose_kernel<<<2048, 256, 0, stream>>>(Ws + (size_t)i * 512 * 1024, 1024, 0,
                                                    512, 1024,
                                                    WsT + (size_t)i * 1024 * 512, 512, 1024);
  convert_pad_kernel<<<4096, 256, 0, stream>>>(inputs, Abf, 32768, 360, 384);
  zero_bf16_kernel<<<64, 256, 0, stream>>>(h, 128 * 512);
  init_sync_kernel<<<1, 1, 0, stream>>>(syncp);

  // --- off-critical-path GEMMs: X = in@W_enc+b ; Xc = X@W0[:300] (all t) ---
  gemm_bf16_kernel<<<5120, 256, 0, stream>>>(Abf, WencT, b_enc, X, 32768, 320, 384, 300);
  gemm_bf16_kernel<<<16384, 256, 0, stream>>>(X, W0topT, nullptr, Xc, 32768, 1024, 320, 1024);

  // --- persistent scan: 1 launch, 256 steps, grid barriers between levels ---
  recurrence_kernel<<<96, 256, 0, stream>>>(Xc, W0hT, WsT, masks, h, Sb, out, syncp);
}
